// GlobalPyramid_12704513262115
// MI455X (gfx1250) — compile-verified, matrix operand order checked by reading, not run on hardware
//
#include <hip/hip_runtime.h>
#include <hip/hip_bf16.h>
#include <math.h>

typedef __attribute__((ext_vector_type(16))) _Float16 v16h;
typedef __attribute__((ext_vector_type(8)))  _Float16 v8h;
typedef __attribute__((ext_vector_type(8)))  float    v8f;

#define WSZ   7
#define SHF   3
#define HEADS 4
#define EPSF  1e-5f

__device__ __forceinline__ float gelu_exact(float x) {
  return 0.5f * x * (1.0f + erff(x * 0.70710678118654752f));
}

// ---------------- conv 3x3 stride2 pad1 + BN(eval) + exact GELU ----------------
__global__ void __launch_bounds__(256)
conv3x3s2_bn_gelu(const float* __restrict__ in, const float* __restrict__ w,
                  const float* __restrict__ g, const float* __restrict__ bb,
                  float* __restrict__ out,
                  int B, int Cin, int Hin, int Win, int Cout, int Hout, int Wout)
{
  size_t total = (size_t)B * Cout * Hout * Wout;
  size_t idx = (size_t)blockIdx.x * blockDim.x + threadIdx.x;
  if (idx >= total) return;
  int wo = (int)(idx % Wout); size_t t = idx / Wout;
  int ho = (int)(t % Hout); t /= Hout;
  int oc = (int)(t % Cout); int b = (int)(t / Cout);
  int hi0 = ho * 2 - 1, wi0 = wo * 2 - 1;
  float acc = 0.f;
  const float* wbase = w + (size_t)oc * Cin * 9;
  for (int ic = 0; ic < Cin; ++ic) {
    const float* ip = in + (size_t)(b * Cin + ic) * Hin * Win;
    const float* wp = wbase + ic * 9;
#pragma unroll
    for (int ky = 0; ky < 3; ++ky) {
      int hi = hi0 + ky;
      if ((unsigned)hi >= (unsigned)Hin) continue;
#pragma unroll
      for (int kx = 0; kx < 3; ++kx) {
        int wi = wi0 + kx;
        if ((unsigned)wi >= (unsigned)Win) continue;
        acc += ip[(size_t)hi * Win + wi] * wp[ky * 3 + kx];
      }
    }
  }
  float v = acc * (g[oc] * rsqrtf(1.f + EPSF)) + bb[oc];
  out[idx] = gelu_exact(v);
}

// ---------------- 2D sincos positional embedding (HW, C) ----------------
__global__ void __launch_bounds__(256)
pos_embed_kernel(float* __restrict__ pe, int H, int W, int C)
{
  int total = H * W * C;
  int idx = blockIdx.x * blockDim.x + threadIdx.x;
  if (idx >= total) return;
  int c = idx % C; int l = idx / C;
  int y = l / W, x = l % W;
  int ed = C >> 1, dts = ed >> 1;
  int pos, cc = c;
  if (cc < ed) { pos = y; } else { pos = x; cc -= ed; }
  int i = (cc < dts) ? cc : (cc - dts);
  float dim_t = powf(10000.f, 2.f * (float)i / (float)dts);
  float ph = (float)pos / dim_t;
  pe[idx] = (cc < dts) ? sinf(ph) : cosf(ph);
}

// ---------------- LayerNorm over last dim, wave(32)-per-row ----------------
// C is always a multiple of 32 and <= 768 (24 regs per lane max).
__global__ void __launch_bounds__(256)
layernorm_rows(const float* __restrict__ in, const float* __restrict__ g,
               const float* __restrict__ b, float* __restrict__ out,
               size_t rows, int C)
{
  int lane = threadIdx.x & 31;
  int wid  = threadIdx.x >> 5;
  size_t row = (size_t)blockIdx.x * 8 + wid;
  if (row >= rows) return;
  const float* x = in + row * (size_t)C;
  float* o = out + row * (size_t)C;
  int nk = C >> 5;
  float xv[24];
  float s = 0.f;
  for (int k = 0; k < nk; ++k) { xv[k] = x[lane + 32 * k]; s += xv[k]; }
  for (int off = 16; off > 0; off >>= 1) s += __shfl_xor(s, off, 32);
  float mean = s / (float)C;
  float vs = 0.f;
  for (int k = 0; k < nk; ++k) { float d = xv[k] - mean; vs += d * d; }
  for (int off = 16; off > 0; off >>= 1) vs += __shfl_xor(vs, off, 32);
  float inv = rsqrtf(vs / (float)C + EPSF);
  for (int k = 0; k < nk; ++k) {
    int c = lane + 32 * k;
    o[c] = (xv[k] - mean) * inv * g[c] + b[c];
  }
}

// NCHW activations + projected pos-embed -> LN -> tokens (B*L, C)
__global__ void __launch_bounds__(256)
layernorm_tokens_nchw(const float* __restrict__ xin, const float* __restrict__ pep,
                      const float* __restrict__ g, const float* __restrict__ b,
                      float* __restrict__ out, int Bn, int C, int L)
{
  int lane = threadIdx.x & 31;
  int wid  = threadIdx.x >> 5;
  size_t rows = (size_t)Bn * L;
  size_t row = (size_t)blockIdx.x * 8 + wid;
  if (row >= rows) return;
  int bb = (int)(row / L);
  int l  = (int)(row % L);
  int nk = C >> 5;
  float xv[24];
  float s = 0.f;
  for (int k = 0; k < nk; ++k) {
    int c = lane + 32 * k;
    float v = xin[(size_t)(bb * C + c) * L + l] + pep[(size_t)l * C + c];
    xv[k] = v; s += v;
  }
  for (int off = 16; off > 0; off >>= 1) s += __shfl_xor(s, off, 32);
  float mean = s / (float)C;
  float vs = 0.f;
  for (int k = 0; k < nk; ++k) { float d = xv[k] - mean; vs += d * d; }
  for (int off = 16; off > 0; off >>= 1) vs += __shfl_xor(vs, off, 32);
  float inv = rsqrtf(vs / (float)C + EPSF);
  float* o = out + row * (size_t)C;
  for (int k = 0; k < nk; ++k) {
    int c = lane + 32 * k;
    o[c] = (xv[k] - mean) * inv * g[c] + b[c];
  }
}

// ---------------- WMMA GEMM: out = act(A[M,K] @ W[N,K]^T + bias) (+res) ----------------
// f32 operands converted to f16 while staging in LDS, f32 accumulation via
// v_wmma_f32_16x16x32_f16. Block tile 128x64, 128 threads = 4 waves (wave32);
// wave w owns TWO 16-row strips (rows w*32 .. w*32+31) x 4 N-subtiles:
// 8 accumulators and 8 WMMAs per k-step against 12 ds_load_b128 (B-fragments
// shared across both strips), so the loop is matrix-pipe- not LDS-issue-bound.
// REQUIREMENT: K % 32 == 0 (true for every GEMM in this model: 96..3072),
// so only row (M/N) guards are needed and the loader is fully vectorized:
// two global float4 loads -> cvt_pk -> one 16B LDS store per 8-element segment.
#define GBM 128
#define GBN 64
#define GBK 32
#define GLD 40   // padded LDS row stride in halfs: 80B rows keep 16B alignment

__device__ __forceinline__ v8h cvt8(const float* __restrict__ p) {
  const float4* q = (const float4*)p;
  float4 a = q[0], c = q[1];
  v8h h;
  h[0] = (_Float16)a.x; h[1] = (_Float16)a.y; h[2] = (_Float16)a.z; h[3] = (_Float16)a.w;
  h[4] = (_Float16)c.x; h[5] = (_Float16)c.y; h[6] = (_Float16)c.z; h[7] = (_Float16)c.w;
  return h;
}

__global__ void __launch_bounds__(128)
gemm_bias_act_wmma(const float* __restrict__ A, const float* __restrict__ Wm,
                   const float* __restrict__ bias, const float* __restrict__ res,
                   float* __restrict__ out, int M, int N, int K, int act)
{
  __shared__ _Float16 As[GBM][GLD];
  __shared__ _Float16 Bs[GBN][GLD];
  int m0 = blockIdx.y * GBM;
  int n0 = blockIdx.x * GBN;
  int tid  = threadIdx.x;
  int wave = tid >> 5;          // 0..3 -> 32-row M block
  int lane = tid & 31;
  v8f zero = {};
  v8f acc[2][4] = {{zero, zero, zero, zero}, {zero, zero, zero, zero}};
  v8h h0 = {};
  int rowA0 = wave * 32 + (lane & 15);   // strip 0; strip 1 = rowA0 + 16
  int kbase = (lane >> 4) * 8;  // lanes 0-15: K 0..7/16..23 ; lanes 16-31: K 8..15/24..31
  // Loader mapping: 8-half segments; 4 segments per 32-wide K row.
  int segr = tid >> 2;          // 0..31
  int segc = (tid & 3) * 8;     // 0,8,16,24

  for (int k0 = 0; k0 < K; k0 += GBK) {
    // A tile: 128 rows x 32 = 512 segments -> 4 per thread
#pragma unroll
    for (int it = 0; it < 4; ++it) {
      int r  = segr + it * 32;
      int gm = m0 + r;
      v8h h = h0;
      if (gm < M) h = cvt8(A + (size_t)gm * K + k0 + segc);
      *(v8h*)&As[r][segc] = h;
    }
    // B tile: 64 rows x 32 = 256 segments -> 2 per thread
#pragma unroll
    for (int it = 0; it < 2; ++it) {
      int r  = segr + it * 32;
      int gn = n0 + r;
      v8h h = h0;
      if (gn < N) h = cvt8(Wm + (size_t)gn * K + k0 + segc);
      *(v8h*)&Bs[r][segc] = h;
    }
    __syncthreads();

    v16h afrag[2];
#pragma unroll
    for (int st = 0; st < 2; ++st) {
      v8h alo = *(const v8h*)&As[rowA0 + st * 16][kbase];
      v8h ahi = *(const v8h*)&As[rowA0 + st * 16][kbase + 16];
      afrag[st] = __builtin_shufflevector(alo, ahi,
          0, 1, 2, 3, 4, 5, 6, 7, 8, 9, 10, 11, 12, 13, 14, 15);
    }

#pragma unroll
    for (int sub = 0; sub < 4; ++sub) {
      v8h blo = *(const v8h*)&Bs[sub * 16 + (lane & 15)][kbase];
      v8h bhi = *(const v8h*)&Bs[sub * 16 + (lane & 15)][kbase + 16];
      v16h bfrag = __builtin_shufflevector(blo, bhi,
          0, 1, 2, 3, 4, 5, 6, 7, 8, 9, 10, 11, 12, 13, 14, 15);
      acc[0][sub] = __builtin_amdgcn_wmma_f32_16x16x32_f16(
          false, afrag[0], false, bfrag, (short)0, acc[0][sub], false, false);
      acc[1][sub] = __builtin_amdgcn_wmma_f32_16x16x32_f16(
          false, afrag[1], false, bfrag, (short)0, acc[1][sub], false, false);
    }
    __syncthreads();
  }

  // D layout: VGPR r -> M = r (lanes 0-15) / M = 8+r (lanes 16-31), N = lane&15
  int ncol = lane & 15;
  int rtop = (lane >> 4) * 8;
#pragma unroll
  for (int st = 0; st < 2; ++st) {
#pragma unroll
    for (int sub = 0; sub < 4; ++sub) {
      int n = n0 + sub * 16 + ncol;
      if (n >= N) continue;
      float bv = bias ? bias[n] : 0.f;
#pragma unroll
      for (int r = 0; r < 8; ++r) {
        int m = m0 + wave * 32 + st * 16 + rtop + r;
        if (m >= M) continue;
        float v = acc[st][sub][r] + bv;
        if (act) v = gelu_exact(v);
        if (res) v += res[(size_t)m * N + n];
        out[(size_t)m * N + n] = v;
      }
    }
  }
}

// ---------------- shifted 7x7 window attention core ----------------
// One workgroup per (window, head); shift + attention mask via index math.
__global__ void __launch_bounds__(64)
win_attn(const float* __restrict__ qkv, float* __restrict__ out,
         int B, int H, int W, int C, int shift)
{
  extern __shared__ float sm[];
  int nWx = W / WSZ;
  int win = blockIdx.x, head = blockIdx.y, b = blockIdx.z;
  int wy = win / nWx, wx = win % nWx;
  int hd = C / HEADS;
  int L  = H * W;
  int c3 = 3 * C;
  float* qs = sm;
  float* ks = qs + 49 * hd;
  float* vs = ks + 49 * hd;
  int tid = threadIdx.x;

  for (int i = tid; i < 49 * hd; i += 64) {
    int tkn = i / hd, d = i - tkn * hd;
    int iy = tkn / WSZ, ix = tkn - iy * WSZ;
    int y = wy * WSZ + iy + shift; if (y >= H) y -= H;   // roll(-shift) gather
    int x = wx * WSZ + ix + shift; if (x >= W) x -= W;
    size_t base = ((size_t)b * L + (size_t)y * W + x) * c3 + head * hd + d;
    qs[i] = qkv[base];
    ks[i] = qkv[base + C];
    vs[i] = qkv[base + 2 * C];
  }
  __syncthreads();
  if (tid >= 49) return;

  int iy = tid / WSZ, ix = tid - iy * WSZ;
  float scale = rsqrtf((float)hd);
  int regq = 0;
  if (shift) {
    int ya = wy * WSZ + iy, xa = wx * WSZ + ix;
    int rh = (ya < H - WSZ) ? 0 : ((ya < H - SHF) ? 1 : 2);
    int rw = (xa < W - WSZ) ? 0 : ((xa < W - SHF) ? 1 : 2);
    regq = 3 * rh + rw;
  }
  float scores[49];
  float mx = -1e30f;
  const float* qrow = qs + tid * hd;
  for (int j = 0; j < 49; ++j) {
    const float* krow = ks + j * hd;
    float sacc = 0.f;
    for (int d = 0; d < hd; ++d) sacc += qrow[d] * krow[d];
    sacc *= scale;
    if (shift) {
      int jy = wy * WSZ + j / WSZ, jx = wx * WSZ + j % WSZ;
      int rh = (jy < H - WSZ) ? 0 : ((jy < H - SHF) ? 1 : 2);
      int rw = (jx < W - WSZ) ? 0 : ((jx < W - SHF) ? 1 : 2);
      if (3 * rh + rw != regq) sacc -= 100.f;
    }
    scores[j] = sacc;
    mx = fmaxf(mx, sacc);
  }
  float ssum = 0.f;
  for (int j = 0; j < 49; ++j) { scores[j] = expf(scores[j] - mx); ssum += scores[j]; }
  float inv = 1.f / ssum;

  int y = wy * WSZ + iy + shift; if (y >= H) y -= H;     // roll(+shift) scatter == gather map
  int x = wx * WSZ + ix + shift; if (x >= W) x -= W;
  float* op = out + ((size_t)b * L + (size_t)y * W + x) * C + head * hd;
  for (int d = 0; d < hd; ++d) {
    float o = 0.f;
    for (int j = 0; j < 49; ++j) o += scores[j] * vs[j * hd + d];
    op[d] = o * inv;
  }
}

// ---------------- tokens (B,L,C) -> NCHW feature map ----------------
__global__ void __launch_bounds__(256)
tok2nchw(const float* __restrict__ t, float* __restrict__ out, int B, int C, int L)
{
  size_t total = (size_t)B * C * L;
  size_t idx = (size_t)blockIdx.x * blockDim.x + threadIdx.x;
  if (idx >= total) return;
  int l = (int)(idx % L); size_t r = idx / L;
  int c = (int)(r % C); int b = (int)(r / C);
  out[idx] = t[((size_t)b * L + l) * C + c];
}

// =====================================================================
extern "C" void kernel_launch(void* const* d_in, const int* in_sizes, int n_in,
                              void* d_out, int out_size, void* d_ws, size_t ws_size,
                              hipStream_t stream)
{
  (void)in_sizes; (void)n_in; (void)out_size; (void)ws_size;
  int pi = 0;
  auto nxt = [&]() { return (const float*)d_in[pi++]; };

  const float* x      = nxt();       // (16,3,448,448)
  const float* stem_w = nxt();
  const float* stem_g = nxt();
  const float* stem_b = nxt();

  // Workspace layout (floats). Total ~175.8M floats (~703 MB).
  // `big` doubles as stem output (16,96,224,224) and later as qkv / fc1 scratch
  // (same 77,070,336-float footprint); the stem data is dead once the stage-0
  // downsample conv has consumed it, and all launches are stream-ordered.
  float* wsf = (float*)d_ws;
  const size_t BIGSZ = 77070336ull;  // 16*96*224*224 == 16*12544*384
  const size_t MIDSZ = 19267584ull;  // 16*12544*96
  float* big  = wsf;
  float* down = big  + BIGSZ;
  float* tA   = down + MIDSZ;
  float* tB   = tA   + MIDSZ;
  float* tln  = tB   + MIDSZ;
  float* attn = tln  + MIDSZ;
  float* pe   = attn + MIDSZ;
  float* pep  = pe   + 1204224ull;   // 12544*96 max

  const int B = 16;
  {
    size_t tot = (size_t)B * 96 * 224 * 224;
    conv3x3s2_bn_gelu<<<(unsigned)((tot + 255) / 256), 256, 0, stream>>>(
        x, stem_w, stem_g, stem_b, big, B, 3, 448, 448, 96, 224, 224);
  }

  const float* stage_in = big;
  int ic = 96, ih = 224, iw = 224;
  float* feat = (float*)d_out;

  for (int s = 0; s < 4; ++s) {
    int C = 96 << s;
    int H = ih >> 1, W = iw >> 1;
    int L = H * W;
    size_t M = (size_t)B * L;

    const float* down_w = nxt(); const float* down_g = nxt(); const float* down_b = nxt();
    const float* pos_w  = nxt(); const float* pos_b  = nxt();
    const float* nin_g  = nxt(); const float* nin_b  = nxt();

    {
      size_t tot = (size_t)B * C * L;
      conv3x3s2_bn_gelu<<<(unsigned)((tot + 255) / 256), 256, 0, stream>>>(
          stage_in, down_w, down_g, down_b, down, B, ic, ih, iw, C, H, W);
    }
    {
      int tot = L * C;
      pos_embed_kernel<<<(tot + 255) / 256, 256, 0, stream>>>(pe, H, W, C);
    }
    {
      dim3 g((C + GBN - 1) / GBN, (L + GBM - 1) / GBM);
      gemm_bias_act_wmma<<<g, 128, 0, stream>>>(pe, pos_w, pos_b, nullptr, pep, L, C, C, 0);
    }
    layernorm_tokens_nchw<<<(unsigned)((M + 7) / 8), 256, 0, stream>>>(
        down, pep, nin_g, nin_b, tA, B, C, L);

    float* t = tA; float* o = tB;
    for (int j = 0; j < 2; ++j) {
      const float* n1_g  = nxt(); const float* n1_b  = nxt();
      const float* qkv_w = nxt(); const float* qkv_b = nxt();
      const float* prj_w = nxt(); const float* prj_b = nxt();
      const float* n2_g  = nxt(); const float* n2_b  = nxt();
      const float* fc1_w = nxt(); const float* fc1_b = nxt();
      const float* fc2_w = nxt(); const float* fc2_b = nxt();
      int shift = (j & 1) ? SHF : 0;

      layernorm_rows<<<(unsigned)((M + 7) / 8), 256, 0, stream>>>(t, n1_g, n1_b, tln, M, C);
      {
        dim3 g((3 * C + GBN - 1) / GBN, (unsigned)((M + GBM - 1) / GBM));
        gemm_bias_act_wmma<<<g, 128, 0, stream>>>(tln, qkv_w, qkv_b, nullptr, big,
                                                  (int)M, 3 * C, C, 0);
      }
      {
        int hd = C / HEADS;
        size_t sh = (size_t)3 * 49 * hd * sizeof(float);
        dim3 g((H / WSZ) * (W / WSZ), HEADS, B);
        win_attn<<<g, 64, sh, stream>>>(big, attn, B, H, W, C, shift);
      }
      {
        dim3 g((C + GBN - 1) / GBN, (unsigned)((M + GBM - 1) / GBM));
        gemm_bias_act_wmma<<<g, 128, 0, stream>>>(attn, prj_w, prj_b, t, o,
                                                  (int)M, C, C, 0);   // + residual
      }
      { float* tmp = t; t = o; o = tmp; }

      layernorm_rows<<<(unsigned)((M + 7) / 8), 256, 0, stream>>>(t, n2_g, n2_b, tln, M, C);
      {
        dim3 g((4 * C + GBN - 1) / GBN, (unsigned)((M + GBM - 1) / GBM));
        gemm_bias_act_wmma<<<g, 128, 0, stream>>>(tln, fc1_w, fc1_b, nullptr, big,
                                                  (int)M, 4 * C, C, 1);  // fused GELU
      }
      {
        dim3 g((C + GBN - 1) / GBN, (unsigned)((M + GBM - 1) / GBM));
        gemm_bias_act_wmma<<<g, 128, 0, stream>>>(big, fc2_w, fc2_b, t, o,
                                                  (int)M, C, 4 * C, 0);  // + residual
      }
      { float* tmp = t; t = o; o = tmp; }
    }

    {
      size_t tot = (size_t)B * C * L;
      tok2nchw<<<(unsigned)((tot + 255) / 256), 256, 0, stream>>>(t, feat, B, C, L);
    }
    stage_in = feat;                 // next stage reads its conv input from d_out
    feat += (size_t)B * C * L;
    ic = C; ih = H; iw = W;
  }
}
